// LcaModule_84516366451281
// MI455X (gfx1250) — matrix operand order repaired and checked by reading, not compiled, not run on hardware
//
#include <hip/hip_runtime.h>

// ---------------------------------------------------------------------------
// LCA sparse-coding inference on MI455X (gfx1250), bf16 WMMA + fp32 accum.
//
//  - total math ~2.3 TFLOP, dominated by 48x (256x8192)@(8192x8192) steps.
//  - Gram matrix in bf16 = 128 MB < 192 MB L2  -> iteration loop is
//    L2-resident; HBM traffic is a one-time ~0.4 GB (negligible @ 23.3 TB/s).
//  - v_wmma_f32_16x16x32_bf16: K=32/instr vs K=4 for f32 WMMA.
//  - K-loop staging uses gfx1250 async copies (global_load_async_to_lds_b128,
//    ASYNCcnt) with double-buffered LDS: issue next tile, compute current,
//    s_wait_asynccnt 0 + one barrier per K-step. No VGPR round-trip, no
//    per-chunk s_wait_loadcnt stalls.
// ---------------------------------------------------------------------------

typedef __attribute__((ext_vector_type(16))) __bf16 v16bf;
typedef __attribute__((ext_vector_type(8)))  float  v8f;

#define BATCH   256
#define NPIX    4096
#define NLAT    8192
#define NSTEPS  50
#define STEPSZ  0.1f
#define LAMBDA  0.1f

static __device__ __forceinline__ unsigned short f32_to_bf16(float f) {
    unsigned int u = __float_as_uint(f);
    unsigned int r = (u + 0x7FFFu + ((u >> 16) & 1u)) >> 16;  // RNE
    return (unsigned short)r;
}

// Low 32 bits of a generic pointer to LDS == workgroup-relative LDS byte addr.
static __device__ __forceinline__ unsigned lds_off32(const void* p) {
    return (unsigned)(unsigned long long)p;
}

// Async 16B copy global -> LDS (VFLAT async op, tracked by ASYNCcnt).
static __device__ __forceinline__ void async_copy_b128(const void* gsrc, unsigned ldsoff) {
    asm volatile("global_load_async_to_lds_b128 %0, %1, off"
                 :: "v"(ldsoff), "v"(gsrc) : "memory");
}

static __device__ __forceinline__ void wait_asynccnt0() {
#if __has_builtin(__builtin_amdgcn_s_wait_asynccnt)
    __builtin_amdgcn_s_wait_asynccnt(0);
#else
    asm volatile("s_wait_asynccnt 0" ::: "memory");
#endif
}

// --------------------------- small prep kernels ----------------------------

__global__ void convert_x_kernel(const float* __restrict__ x,
                                 unsigned short* __restrict__ xb, int n) {
    int i = blockIdx.x * blockDim.x + threadIdx.x;
    if (i < n) xb[i] = f32_to_bf16(x[i]);
}

// w [NPIX][NLAT] f32 -> wb [NPIX][NLAT] bf16 and wt [NLAT][NPIX] bf16
__global__ void convert_transpose_w_kernel(const float* __restrict__ w,
                                           unsigned short* __restrict__ wb,
                                           unsigned short* __restrict__ wt) {
    __shared__ unsigned short tile[32][33];
    const int tx = threadIdx.x, ty = threadIdx.y;
    const int p0 = blockIdx.x * 32;
    const int l0 = blockIdx.y * 32;
#pragma unroll
    for (int i = 0; i < 4; ++i) {
        int p = p0 + ty + i * 8;
        size_t idx = (size_t)p * NLAT + (l0 + tx);
        unsigned short h = f32_to_bf16(w[idx]);
        wb[idx] = h;
        tile[ty + i * 8][tx] = h;
    }
    __syncthreads();
#pragma unroll
    for (int i = 0; i < 4; ++i) {
        int l = l0 + ty + i * 8;
        wt[(size_t)l * NPIX + (p0 + tx)] = tile[tx][ty + i * 8];
    }
}

__global__ void threshold_kernel(const float* __restrict__ u,
                                 unsigned short* __restrict__ a, int n) {
    int i = blockIdx.x * blockDim.x + threadIdx.x;
    if (i < n) {
        float v = u[i];
        float t = (v > LAMBDA) ? (v - LAMBDA) : 0.0f;
        a[i] = f32_to_bf16(t);
    }
}

// ------------------------------ WMMA NT GEMM -------------------------------
// C[M x N] = A[M x K] * Bt[N x K]^T   (both bf16 row-major, f32 accumulate)
// MODE 0: out0 = C (excit), out1 = 0.1*C (first u update from u0 = 0)
// MODE 1: outh = bf16(C - I)          (Gram with -I baked in; symmetric)
// MODE 2: out0(u) = 0.9*u + 0.1*(aux(excit) - C)     (LCA step)
// MODE 3: out0 = C                    (reconstruction)

template <int MODE>
__global__ __launch_bounds__(256)
void gemm_nt_kernel(const unsigned short* __restrict__ A, int lda,
                    const unsigned short* __restrict__ Bt, int ldb,
                    int K,
                    float* __restrict__ out0,
                    float* __restrict__ out1,
                    const float* __restrict__ aux,
                    unsigned short* __restrict__ outh,
                    int ldc) {
    constexpr int BM = 128, BN = 128, BK = 32;
    constexpr int RSTR  = 80;                 // 64B data + 16B pad per LDS row
    constexpr int BUFSZ = (BM + BN) * RSTR;   // 20480 B per stage
    __shared__ __align__(16) unsigned char lds[2 * BUFSZ];  // 40 KB << 320 KB/WGP

    const int tid  = threadIdx.x;
    const int wid  = tid >> 5;
    const int lane = tid & 31;
    const int half = lane >> 4;
    const int lrow = lane & 15;
    const int wm   = wid & 3;     // 4 waves across M (32 rows each)
    const int wn   = wid >> 2;    // 2 waves across N (64 cols each)
    const long m0  = (long)blockIdx.y * BM;
    const long n0  = (long)blockIdx.x * BN;

    // Per-thread staging assignment: 2 A chunks + 2 B chunks of 16 B each.
    const int rA0 = tid >> 2,         cA0 = tid & 3;
    const int rA1 = (tid + 256) >> 2, cA1 = tid & 3;
    const int rB0 = rA0, cB0 = cA0;
    const int rB1 = rA1, cB1 = cA1;
    const unsigned short* gA0 = A  + (size_t)(m0 + rA0) * lda + cA0 * 8;
    const unsigned short* gA1 = A  + (size_t)(m0 + rA1) * lda + cA1 * 8;
    const unsigned short* gB0 = Bt + (size_t)(n0 + rB0) * ldb + cB0 * 8;
    const unsigned short* gB1 = Bt + (size_t)(n0 + rB1) * ldb + cB1 * 8;
    const unsigned oA0 = lds_off32(lds + rA0 * RSTR + cA0 * 16);
    const unsigned oA1 = lds_off32(lds + rA1 * RSTR + cA1 * 16);
    const unsigned oB0 = lds_off32(lds + BM * RSTR + rB0 * RSTR + cB0 * 16);
    const unsigned oB1 = lds_off32(lds + BM * RSTR + rB1 * RSTR + cB1 * 16);

    auto issue_tile = [&](int k0, int buf) {
        const unsigned boff = (unsigned)buf * BUFSZ;
        async_copy_b128(gA0 + k0, oA0 + boff);
        async_copy_b128(gA1 + k0, oA1 + boff);
        async_copy_b128(gB0 + k0, oB0 + boff);
        async_copy_b128(gB1 + k0, oB1 + boff);
        // Speculative prefetch of the tile after next (dropped if invalid).
        __builtin_prefetch(gA0 + k0 + BK, 0, 1);
        __builtin_prefetch(gB0 + k0 + BK, 0, 1);
    };

    v8f acc[2][4];
#pragma unroll
    for (int mi = 0; mi < 2; ++mi)
#pragma unroll
        for (int ni = 0; ni < 4; ++ni) acc[mi][ni] = (v8f){0,0,0,0,0,0,0,0};

    issue_tile(0, 0);
    wait_asynccnt0();
    __syncthreads();

    const int nk = K / BK;
    for (int t = 0; t < nk; ++t) {
        const int cur = t & 1;
        if (t + 1 < nk) issue_tile((t + 1) * BK, cur ^ 1);  // overlap with compute

        unsigned char* As = lds + cur * BUFSZ;
        unsigned char* Bs = As + BM * RSTR;

        // A fragment (16x32): lane<16 -> K{0..7}u{16..23}; lane>=16 -> K{8..15}u{24..31}
        v16bf afrag[2];
#pragma unroll
        for (int mi = 0; mi < 2; ++mi) {
            int row = wm * 32 + mi * 16 + lrow;
            union { uint4 q[2]; v16bf v; } ua;
            ua.q[0] = *(const uint4*)(As + row * RSTR + half * 16);
            ua.q[1] = *(const uint4*)(As + row * RSTR + 32 + half * 16);
            afrag[mi] = ua.v;
        }
        // B fragment (32x16): lane<16 -> K{0..7}u{16..23}; lane>=16 -> K{8..15}u{24..31} (col = lrow)
        v16bf bfrag[4];
#pragma unroll
        for (int ni = 0; ni < 4; ++ni) {
            int row = wn * 64 + ni * 16 + lrow;
            union { uint4 q[2]; v16bf v; } ub;
            ub.q[0] = *(const uint4*)(Bs + row * RSTR + half * 16);
            ub.q[1] = *(const uint4*)(Bs + row * RSTR + 32 + half * 16);
            bfrag[ni] = ub.v;
        }
#pragma unroll
        for (int mi = 0; mi < 2; ++mi)
#pragma unroll
            for (int ni = 0; ni < 4; ++ni)
                acc[mi][ni] = __builtin_amdgcn_wmma_f32_16x16x32_bf16(
                    false, afrag[mi], false, bfrag[ni],
                    (short)0, acc[mi][ni], false, false);

        wait_asynccnt0();   // next tile's copies landed in LDS
        __syncthreads();    // all waves done reading current buffer
    }

    // Epilogue. C fragment layout: VGPR v -> row (v or v+8), col = lrow.
#pragma unroll
    for (int mi = 0; mi < 2; ++mi) {
#pragma unroll
        for (int ni = 0; ni < 4; ++ni) {
#pragma unroll
            for (int v = 0; v < 8; ++v) {
                long r = m0 + wm * 32 + mi * 16 + (half ? v + 8 : v);
                long c = n0 + wn * 64 + ni * 16 + lrow;
                size_t idx = (size_t)r * (size_t)ldc + (size_t)c;
                float val = acc[mi][ni][v];
                if constexpr (MODE == 0) {
                    out0[idx] = val;            // excit
                    out1[idx] = STEPSZ * val;   // u after first update (u0=0)
                } else if constexpr (MODE == 1) {
                    outh[idx] = f32_to_bf16(val - (r == c ? 1.0f : 0.0f));
                } else if constexpr (MODE == 2) {
                    float uo = out0[idx];
                    out0[idx] = (1.0f - STEPSZ) * uo + STEPSZ * (aux[idx] - val);
                } else {
                    out0[idx] = val;
                }
            }
        }
    }
}

// ------------------------------- launcher ----------------------------------

extern "C" void kernel_launch(void* const* d_in, const int* in_sizes, int n_in,
                              void* d_out, int out_size, void* d_ws, size_t ws_size,
                              hipStream_t stream) {
    (void)in_sizes; (void)n_in; (void)out_size; (void)ws_size;
    const float* x = (const float*)d_in[0];   // [BATCH, NPIX]
    const float* w = (const float*)d_in[1];   // [NPIX, NLAT]
    float* out = (float*)d_out;               // [BATCH, NPIX]

    char* ws = (char*)d_ws;
    size_t off = 0;
    auto alloc = [&](size_t bytes) -> void* {
        void* p = ws + off;
        off = (off + bytes + 255) & ~(size_t)255;
        return p;
    };
    unsigned short* wb    = (unsigned short*)alloc((size_t)NPIX * NLAT * 2);  //  64 MB
    unsigned short* wt    = (unsigned short*)alloc((size_t)NLAT * NPIX * 2);  //  64 MB
    unsigned short* gram  = (unsigned short*)alloc((size_t)NLAT * NLAT * 2);  // 128 MB (fits L2)
    unsigned short* xb    = (unsigned short*)alloc((size_t)BATCH * NPIX * 2);
    unsigned short* ab    = (unsigned short*)alloc((size_t)BATCH * NLAT * 2);
    float*          excit = (float*)alloc((size_t)BATCH * NLAT * 4);
    float*          u     = (float*)alloc((size_t)BATCH * NLAT * 4);

    // 1) precision conversion + transpose of dictionary
    convert_x_kernel<<<(BATCH * NPIX) / 256, 256, 0, stream>>>(x, xb, BATCH * NPIX);
    convert_transpose_w_kernel<<<dim3(NPIX / 32, NLAT / 32), dim3(32, 8), 0, stream>>>(w, wb, wt);

    // 2) excit = x @ w  (B fragments = rows of wt); fused first u-update
    gemm_nt_kernel<0><<<dim3(NLAT / 128, BATCH / 128), 256, 0, stream>>>(
        xb, NPIX, wt, NPIX, NPIX, excit, u, nullptr, nullptr, NLAT);

    // 3) G = w^T w - I  (A = wt rows, B fragments = rows of wt), bf16 output
    gemm_nt_kernel<1><<<dim3(NLAT / 128, NLAT / 128), 256, 0, stream>>>(
        wt, NPIX, wt, NPIX, NPIX, nullptr, nullptr, nullptr, gram, NLAT);

    // 4) remaining 48 LCA updates; G symmetric so B fragments = rows of G
    for (int s = 0; s < NSTEPS - 2; ++s) {
        threshold_kernel<<<(BATCH * NLAT) / 256, 256, 0, stream>>>(u, ab, BATCH * NLAT);
        gemm_nt_kernel<2><<<dim3(NLAT / 128, BATCH / 128), 256, 0, stream>>>(
            ab, NLAT, gram, NLAT, NLAT, u, nullptr, excit, nullptr, NLAT);
    }

    // 5) out = threshold(u) @ w^T  (B fragments = rows of wb)
    threshold_kernel<<<(BATCH * NLAT) / 256, 256, 0, stream>>>(u, ab, BATCH * NLAT);
    gemm_nt_kernel<3><<<dim3(NPIX / 128, BATCH / 128), 256, 0, stream>>>(
        ab, NLAT, wb, NLAT, NLAT, out, nullptr, nullptr, nullptr, NPIX);
}
